// SpectralStructuralLoss_87540023427756
// MI455X (gfx1250) — matrix operand order repaired and checked by reading, not compiled, not run on hardware
//
#include <hip/hip_runtime.h>
#include <math.h>

// ---------------------------------------------------------------------------
// Spectral + structural pansharpening loss for MI455X (gfx1250, wave32).
// Rank-1 factorized 41x41 MTF convs as banded-matrix WMMA
// (v_wmma_f32_16x16x32_f16, f32 accumulate).  Banded tap matrices are
// precomputed in the exact WMMA VGPR lane layout; image tiles are staged in
// LDS pre-permuted as f16 so operands load as contiguous b128s.
// ---------------------------------------------------------------------------

#define HH 512
#define WW 512
#define NB 8
#define KS 41
#define PADK 20
#define EPSF 1e-20f
#define HWW (HH * WW)

typedef _Float16 v16h __attribute__((ext_vector_type(16)));
typedef float    v8f  __attribute__((ext_vector_type(8)));

__device__ __forceinline__ v8f wmma16(v16h a, v16h b, v8f c) {
  return __builtin_amdgcn_wmma_f32_16x16x32_f16(false, a, false, b, (short)0, c,
                                                false, false);
}
__device__ __forceinline__ int clampi(int x, int lo, int hi) {
  return x < lo ? lo : (x > hi ? hi : x);
}

// ---------------------------------------------------------------------------
// K0: rank-1 factorization of the 8 MTF kernels + zero accumulators.
// uv: band*128 + i -> u (vertical taps); band*128 + 64 + j -> v (horizontal).
// ---------------------------------------------------------------------------
__global__ void k_factor(const float* __restrict__ kern, float* __restrict__ uv,
                         float* __restrict__ acc) {
  int t = threadIdx.x;
  if (t < 8) acc[t] = 0.0f;
  for (int i = t; i < NB * KS; i += blockDim.x) {
    int band = i / KS, j = i % KS;
    const float* kb = kern + band * KS * KS;
    float s = sqrtf(kb[PADK * KS + PADK]);
    uv[band * 128 + j]      = kb[j * KS + PADK] / s;
    uv[band * 128 + 64 + j] = kb[PADK * KS + j] / s;
  }
}

// ---------------------------------------------------------------------------
// K0b: precompute banded tap matrices in WMMA per-lane layout (f16).
//  WH [8][4][32]v16h : B-operand, v taps, output-stride 4 (spec horizontal)
//  WV [8][4][32]v16h : A-operand, u taps, output-stride 4 (spec vertical)
//  PH [2][32]v16h    : B-operand, band-0 v taps, stride 1 (pan horizontal)
//  PV [2][32]v16h    : A-operand, band-0 u taps, stride 1 (pan vertical)
// A lane layout: M=lane&15, K=((lane>>4)<<4)+e
// B lane layout: N=lane&15, K=(e&7)+((e>>3)<<4)+((lane>>4)<<3)
// ---------------------------------------------------------------------------
__global__ void k_bands(const float* __restrict__ uv, _Float16* __restrict__ bm) {
  int tid = threadIdx.x;
  for (int idx = tid; idx < NB * 4 * 32 * 16; idx += blockDim.x) {
    int e = idx & 15, lane = (idx >> 4) & 31, ch = (idx >> 9) & 3, c = idx >> 11;
    int n = lane & 15;
    int Kb = ch * 32 + (e & 7) + ((e >> 3) << 4) + ((lane >> 4) << 3);
    int tp = Kb - 4 * n;
    bm[idx] = (tp >= 0 && tp < KS) ? (_Float16)uv[c * 128 + 64 + tp]
                                   : (_Float16)0.0f;
    int Ka = ch * 32 + ((lane >> 4) << 4) + e;
    int ta = Ka - 4 * n;
    bm[16384 + idx] = (ta >= 0 && ta < KS) ? (_Float16)uv[c * 128 + ta]
                                           : (_Float16)0.0f;
  }
  for (int idx = tid; idx < 2 * 32 * 16; idx += blockDim.x) {
    int e = idx & 15, lane = (idx >> 4) & 31, ch = idx >> 9;
    int n = lane & 15;
    int Kb = ch * 32 + (e & 7) + ((e >> 3) << 4) + ((lane >> 4) << 3);
    int tp = Kb - n;
    bm[32768 + idx] = (tp >= 0 && tp < KS) ? (_Float16)uv[64 + tp]
                                           : (_Float16)0.0f;
    int Ka = ch * 32 + ((lane >> 4) << 4) + e;
    int ta = Ka - n;
    bm[33792 + idx] = (ta >= 0 && ta < KS) ? (_Float16)uv[ta] : (_Float16)0.0f;
  }
}

// ---------------------------------------------------------------------------
// K1: horizontal MTF conv of outputs at subsampled cols (stride 4, offset 2),
// zero pad.  t_h[c][y][xo] = sum_t outputs[c][y][4*xo-18+t] * v[t]
// ---------------------------------------------------------------------------
__global__ void k_spec_h(const float* __restrict__ img,
                         const v16h* __restrict__ WH,
                         float* __restrict__ th) {
  int c = blockIdx.z, y0 = blockIdx.y * 16, xo0 = blockIdx.x * 16;
  int lane = threadIdx.x & 31;
  __shared__ v16h sAh[4][32];
  _Float16* sa = (_Float16*)sAh;
  const float* src = img + c * HWW + y0 * WW;
  int base = 4 * xo0 - 18;
  for (int idx = lane; idx < 16 * 128; idx += 32) {
    int r = idx >> 7, cc = idx & 127;
    int ic = base + cc;
    int icc = clampi(ic, 0, WW - 1);
    float v = src[r * WW + icc];
    v = (ic == icc) ? v : 0.0f;                       // zero pad, branch-free
    int ch = cc >> 5, Kl = cc & 31;
    int dl = r + ((Kl >> 3) & 1) * 16;                // A permutation
    int e = (Kl & 7) | ((Kl & 16) >> 1);
    sa[(ch * 32 + dl) * 16 + e] = (_Float16)v;
  }
  __syncthreads();
  const v16h* wh = WH + c * 4 * 32;
  v8f d = {};
#pragma unroll
  for (int ch = 0; ch < 4; ++ch)
    d = wmma16(sAh[ch][lane], wh[ch * 32 + lane], d);
  float* dst = th + c * HH * 128;
  int lm = lane & 15, lhi = lane >> 4;
#pragma unroll
  for (int r = 0; r < 8; ++r)
    dst[(y0 + r + lhi * 8) * 128 + xo0 + lm] = d[r];
}

// ---------------------------------------------------------------------------
// K2: vertical conv at subsampled rows + masked-L1 partial reduction.
// ---------------------------------------------------------------------------
__global__ void k_spec_v(const float* __restrict__ th,
                         const v16h* __restrict__ WV,
                         const float* __restrict__ lab_b,
                         const float* __restrict__ mask,
                         float* __restrict__ acc) {
  int c = blockIdx.z, yo0 = blockIdx.y * 16, xo0 = blockIdx.x * 16;
  int lane = threadIdx.x & 31;
  __shared__ v16h sBh[4][32];
  _Float16* sb = (_Float16*)sBh;
  int row_base = 4 * yo0 - 18;
  const float* tc = th + c * HH * 128;
  for (int idx = lane; idx < 128 * 16; idx += 32) {
    int Kp = idx >> 4, n = idx & 15;
    int row = row_base + Kp;
    int rc = clampi(row, 0, HH - 1);
    float v = tc[rc * 128 + xo0 + n];
    v = (row == rc) ? v : 0.0f;
    int ch = Kp >> 5, Kl = Kp & 31;
    int dl = n + ((Kl >> 4) << 4);                    // B permutation
    int e = Kl & 15;
    sb[(ch * 32 + dl) * 16 + e] = (_Float16)v;
  }
  __syncthreads();
  const v16h* wv = WV + c * 4 * 32;
  v8f d = {};
#pragma unroll
  for (int ch = 0; ch < 4; ++ch)
    d = wmma16(wv[ch * 32 + lane], sBh[ch][lane], d);
  const float* lab = lab_b + c * HWW;
  const float* mk  = mask  + c * HWW;
  int lm = lane & 15, lhi = lane >> 4;
  float lsum = 0.0f, msum = 0.0f;
#pragma unroll
  for (int r = 0; r < 8; ++r) {
    int iy = 4 * (yo0 + r + lhi * 8) + 2, ix = 4 * (xo0 + lm) + 2;
    float m = mk[iy * WW + ix];
    lsum += m * fabsf(d[r] - lab[iy * WW + ix]);
    msum += m;
  }
  for (int off = 16; off > 0; off >>= 1) {
    lsum += __shfl_down(lsum, off, 32);
    msum += __shfl_down(msum, off, 32);
  }
  if (lane == 0) { atomicAdd(acc + 0, lsum); atomicAdd(acc + 1, msum); }
}

// ---------------------------------------------------------------------------
// K3: dense horizontal conv of PAN (edge pad), stride 1.
// ---------------------------------------------------------------------------
__global__ void k_pan_h(const float* __restrict__ pan,
                        const v16h* __restrict__ PH,
                        float* __restrict__ tmp) {
  int y0 = blockIdx.y * 16, x0 = blockIdx.x * 16;
  int lane = threadIdx.x & 31;
  __shared__ v16h sAh[2][32];
  _Float16* sa = (_Float16*)sAh;
  int base = x0 - PADK;
  for (int idx = lane; idx < 16 * 64; idx += 32) {
    int r = idx >> 6, cc = idx & 63;
    int ic = clampi(base + cc, 0, WW - 1);            // edge pad
    float v = pan[(y0 + r) * WW + ic];
    int ch = cc >> 5, Kl = cc & 31;
    int dl = r + ((Kl >> 3) & 1) * 16;
    int e = (Kl & 7) | ((Kl & 16) >> 1);
    sa[(ch * 32 + dl) * 16 + e] = (_Float16)v;
  }
  __syncthreads();
  v8f d = {};
#pragma unroll
  for (int ch = 0; ch < 2; ++ch)
    d = wmma16(sAh[ch][lane], PH[ch * 32 + lane], d);
  int lm = lane & 15, lhi = lane >> 4;
#pragma unroll
  for (int r = 0; r < 8; ++r)
    tmp[(y0 + r + lhi * 8) * WW + x0 + lm] = d[r];
}

// ---------------------------------------------------------------------------
// K4: dense vertical conv of PAN (edge pad), stride 1.  A = banded u.
// ---------------------------------------------------------------------------
__global__ void k_pan_v(const float* __restrict__ tmp,
                        const v16h* __restrict__ PV,
                        float* __restrict__ panf) {
  int y0 = blockIdx.y * 16, x0 = blockIdx.x * 16;
  int lane = threadIdx.x & 31;
  __shared__ v16h sBh[2][32];
  _Float16* sb = (_Float16*)sBh;
  for (int idx = lane; idx < 64 * 16; idx += 32) {
    int Kp = idx >> 4, n = idx & 15;
    int row = clampi(y0 - PADK + Kp, 0, HH - 1);      // edge pad
    float v = tmp[row * WW + x0 + n];
    int ch = Kp >> 5, Kl = Kp & 31;
    int dl = n + ((Kl >> 4) << 4);
    int e = Kl & 15;
    sb[(ch * 32 + dl) * 16 + e] = (_Float16)v;
  }
  __syncthreads();
  v8f d = {};
#pragma unroll
  for (int ch = 0; ch < 2; ++ch)
    d = wmma16(PV[ch * 32 + lane], sBh[ch][lane], d);
  int lm = lane & 15, lhi = lane >> 4;
#pragma unroll
  for (int r = 0; r < 8; ++r)
    panf[(y0 + r + lhi * 8) * WW + x0 + lm] = d[r];
}

// ---------------------------------------------------------------------------
// K5: dst = src - boxsum(src, w)/(4w^2), zero pad, separable sums in LDS.
// ---------------------------------------------------------------------------
__global__ void k_mean(const float* __restrict__ src, float* __restrict__ dst,
                       int w) {
  int c = blockIdx.z;
  int y0 = blockIdx.y * 16, x0 = blockIdx.x * 16;
  int tid = threadIdx.x;
  int S = 2 * w + 15;
  int oy = y0 + 1 - w, ox = x0 + 1 - w;
  __shared__ float tile[31 * 32];
  __shared__ float rs[31 * 16];
  const float* im = src + c * HWW;
  for (int idx = tid; idx < S * S; idx += 256) {
    int r = idx / S, cc = idx - r * S;
    int gy = oy + r, gx = ox + cc;
    int cy = clampi(gy, 0, HH - 1), cx = clampi(gx, 0, WW - 1);
    float v = im[cy * WW + cx];
    tile[r * 32 + cc] = (gy == cy && gx == cx) ? v : 0.0f;
  }
  __syncthreads();
  for (int idx = tid; idx < S * 16; idx += 256) {
    int r = idx >> 4, j = idx & 15;
    float s = 0.0f;
    for (int d = 0; d < 2 * w; ++d) s += tile[r * 32 + j + d];
    rs[r * 16 + j] = s;
  }
  __syncthreads();
  int i = tid >> 4, j = tid & 15;
  float s = 0.0f;
  for (int d = 0; d < 2 * w; ++d) s += rs[(i + d) * 16 + j];
  dst[c * HWW + (y0 + i) * WW + x0 + j] =
      im[(y0 + i) * WW + x0 + j] - s / (float)(4 * w * w);
}

// ---------------------------------------------------------------------------
// K6: fused structural loss (both xcorrs, threshold, block reduction).
// ---------------------------------------------------------------------------
__global__ void k_final(const float* __restrict__ da_out,
                        const float* __restrict__ db_pan,
                        const float* __restrict__ da_ms,
                        const float* __restrict__ db_panf,
                        float* __restrict__ acc) {
  int y0 = blockIdx.y * 16, x0 = blockIdx.x * 16;
  int tid = threadIdx.x;
  int i = tid >> 4, j = tid & 15;
  __shared__ float t8a[31 * 32], t8b[31 * 32];
  __shared__ float t2a[19 * 20], t2b[19 * 20];
  __shared__ float rs8ab[31 * 16], rs8aa[31 * 16], rs8bb[31 * 16];
  __shared__ float rs2ab[19 * 16], rs2aa[19 * 16], rs2bb[19 * 16];
  __shared__ float red[256];

  for (int idx = tid; idx < 31 * 31; idx += 256) {
    int r = idx / 31, cc = idx - r * 31;
    int gy = y0 - 7 + r, gx = x0 - 7 + cc;
    int cy = clampi(gy, 0, HH - 1), cx = clampi(gx, 0, WW - 1);
    float v = db_panf[cy * WW + cx];
    t8b[r * 32 + cc] = (gy == cy && gx == cx) ? v : 0.0f;
  }
  for (int idx = tid; idx < 19 * 19; idx += 256) {
    int r = idx / 19, cc = idx - r * 19;
    int gy = y0 - 1 + r, gx = x0 - 1 + cc;
    int cy = clampi(gy, 0, HH - 1), cx = clampi(gx, 0, WW - 1);
    float v = db_pan[cy * WW + cx];
    t2b[r * 20 + cc] = (gy == cy && gx == cx) ? v : 0.0f;
  }
  __syncthreads();
  for (int idx = tid; idx < 31 * 16; idx += 256) {
    int r = idx >> 4, jj = idx & 15;
    float s = 0.0f;
    for (int d = 0; d < 16; ++d) { float v = t8b[r * 32 + jj + d]; s += v * v; }
    rs8bb[idx] = s;
  }
  for (int idx = tid; idx < 19 * 16; idx += 256) {
    int r = idx >> 4, jj = idx & 15;
    float s = 0.0f;
    for (int d = 0; d < 4; ++d) { float v = t2b[r * 20 + jj + d]; s += v * v; }
    rs2bb[idx] = s;
  }
  __syncthreads();

  float ysum = 0.0f;
  for (int c = 0; c < NB; ++c) {
    const float* am = da_ms + c * HWW;
    const float* ao = da_out + c * HWW;
    if (c + 1 < NB)
      __builtin_prefetch(da_ms + (c + 1) * HWW + y0 * WW + x0, 0, 1);
    for (int idx = tid; idx < 31 * 31; idx += 256) {
      int r = idx / 31, cc = idx - r * 31;
      int gy = y0 - 7 + r, gx = x0 - 7 + cc;
      int cy = clampi(gy, 0, HH - 1), cx = clampi(gx, 0, WW - 1);
      float v = am[cy * WW + cx];
      t8a[r * 32 + cc] = (gy == cy && gx == cx) ? v : 0.0f;
    }
    for (int idx = tid; idx < 19 * 19; idx += 256) {
      int r = idx / 19, cc = idx - r * 19;
      int gy = y0 - 1 + r, gx = x0 - 1 + cc;
      int cy = clampi(gy, 0, HH - 1), cx = clampi(gx, 0, WW - 1);
      float v = ao[cy * WW + cx];
      t2a[r * 20 + cc] = (gy == cy && gx == cx) ? v : 0.0f;
    }
    __syncthreads();
    for (int idx = tid; idx < 31 * 16; idx += 256) {
      int r = idx >> 4, jj = idx & 15;
      float sab = 0.0f, saa = 0.0f;
      for (int d = 0; d < 16; ++d) {
        float va = t8a[r * 32 + jj + d];
        sab += va * t8b[r * 32 + jj + d];
        saa += va * va;
      }
      rs8ab[idx] = sab; rs8aa[idx] = saa;
    }
    for (int idx = tid; idx < 19 * 16; idx += 256) {
      int r = idx >> 4, jj = idx & 15;
      float sab = 0.0f, saa = 0.0f;
      for (int d = 0; d < 4; ++d) {
        float va = t2a[r * 20 + jj + d];
        sab += va * t2b[r * 20 + jj + d];
        saa += va * va;
      }
      rs2ab[idx] = sab; rs2aa[idx] = saa;
    }
    __syncthreads();
    float s8ab = 0.f, s8aa = 0.f, s8bb = 0.f;
    for (int d = 0; d < 16; ++d) {
      int r = (i + d) * 16 + j;
      s8ab += rs8ab[r]; s8aa += rs8aa[r]; s8bb += rs8bb[r];
    }
    float s2ab = 0.f, s2aa = 0.f, s2bb = 0.f;
    for (int d = 0; d < 4; ++d) {
      int r = (i + d) * 16 + j;
      s2ab += rs2ab[r]; s2aa += rs2aa[r]; s2bb += rs2bb[r];
    }
    float thr = 1.0f - s8ab / (sqrtf(s8aa * s8bb) + EPSF);
    float Xc = fmaxf(s2ab / (sqrtf(s2aa * s2bb) + EPSF), -1.0f);
    float X = 1.0f - Xc;
    ysum += (X > thr) ? X : 0.0f;
    __syncthreads();
  }
  red[tid] = ysum;
  __syncthreads();
  for (int s = 128; s > 0; s >>= 1) {
    if (tid < s) red[tid] += red[tid + s];
    __syncthreads();
  }
  if (tid == 0) atomicAdd(acc + 2, red[0]);
}

// K7: L = L_spec + BETA * mean(Y)
__global__ void k_out(const float* __restrict__ acc, float* __restrict__ out) {
  out[0] = acc[0] / acc[1] + 0.25f * acc[2] / 8388608.0f;  // B*NB*H*W
}

extern "C" void kernel_launch(void* const* d_in, const int* in_sizes, int n_in,
                              void* d_out, int out_size, void* d_ws,
                              size_t ws_size, hipStream_t stream) {
  const float* outputs = (const float*)d_in[0];   // (4,8,512,512)
  const float* labels  = (const float*)d_in[1];   // (4,9,512,512)
  const float* inp     = (const float*)d_in[2];   // (4,9,512,512)
  const float* mtf     = (const float*)d_in[3];   // (8,1,41,41)
  const float* mask    = (const float*)d_in[4];   // (1,8,512,512)
  float* ws = (float*)d_ws;

  float* acc = ws;                               // 8 floats
  float* uv  = ws + 8;                           // 1024 floats
  _Float16* bm = (_Float16*)(ws + 1536);         // 34816 halves (17408 floats)
  const v16h* WH = (const v16h*)bm;              // [8][4][32]
  const v16h* WV = (const v16h*)(bm + 16384);    // [8][4][32]
  const v16h* PH = (const v16h*)(bm + 32768);    // [2][32]
  const v16h* PV = (const v16h*)(bm + 33792);    // [2][32]
  float* t_h     = ws + 20480;                   // 8*512*128
  float* tmpP    = t_h + NB * HH * 128;          // 512*512
  float* panf    = tmpP + HWW;                   // 512*512
  float* da_out  = panf + HWW;                   // 8*512*512
  float* da_ms   = da_out + NB * HWW;            // 8*512*512
  float* db_pan  = da_ms + NB * HWW;             // 512*512
  float* db_panf = db_pan + HWW;                 // 512*512

  k_factor<<<1, 64, 0, stream>>>(mtf, uv, acc);
  k_bands<<<1, 256, 0, stream>>>(uv, bm);

  for (int b = 0; b < 4; ++b) {
    const float* out_b   = outputs + (size_t)b * NB * HWW;
    const float* lab_b   = labels  + (size_t)b * (NB + 1) * HWW;
    const float* inp_b   = inp     + (size_t)b * (NB + 1) * HWW;
    const float* pan_inp = inp_b + NB * HWW;
    const float* pan_lab = lab_b + NB * HWW;

    k_spec_h<<<dim3(8, 32, NB), 32, 0, stream>>>(out_b, WH, t_h);
    k_spec_v<<<dim3(8, 8, NB), 32, 0, stream>>>(t_h, WV, lab_b, mask, acc);

    k_pan_h<<<dim3(32, 32), 32, 0, stream>>>(pan_inp, PH, tmpP);
    k_pan_v<<<dim3(32, 32), 32, 0, stream>>>(tmpP, PV, panf);

    k_mean<<<dim3(32, 32, NB), 256, 0, stream>>>(out_b, da_out, 2);
    k_mean<<<dim3(32, 32, 1), 256, 0, stream>>>(pan_lab, db_pan, 2);
    k_mean<<<dim3(32, 32, NB), 256, 0, stream>>>(inp_b, da_ms, 8);
    k_mean<<<dim3(32, 32, 1), 256, 0, stream>>>(panf, db_panf, 8);

    k_final<<<dim3(32, 32), 256, 0, stream>>>(da_out, db_pan, da_ms, db_panf,
                                              acc);
  }
  k_out<<<1, 1, 0, stream>>>(acc, (float*)d_out);
}
